// ImputationLoss_1159641169955
// MI455X (gfx1250) — matrix operand order repaired and checked by reading, not compiled, not run on hardware
//
#include <hip/hip_runtime.h>
#include <hip/hip_bf16.h>
#include <cstdint>

#define B_   32
#define L_   8192
#define BL_  (B_ * L_)
#define JS_EPS_F 1e-7f
#define EPS_F    1e-8f
#define EVO_LAMBDA_F 3.0f
#define EVO_WEIGHT_F 50.0f
#define KSPLIT 16

typedef __attribute__((ext_vector_type(16))) _Float16 v16h;
typedef __attribute__((ext_vector_type(8)))  _Float16 v8h;
typedef __attribute__((ext_vector_type(8)))  float    v8f;

// ---- workspace layout (bytes) ----
#define OFF_CEPART 0u          // 1024 floats
#define OFF_R2PART 4096u       // 256 floats
#define OFF_CNT    5120u       // 5*32*32 floats (planes: class0..3, valid)
#define OFF_SSAME  25600u      // 32*32 floats
#define OFF_SDIFF  29696u      // 32*32 floats
#define OFF_PLOGP  33792u      // 32*8192 floats
#define OFF_LABEL  1082368u    // 32*8192 bytes
#define OFF_XIND   1344512u    // 5*32*8192 halves (16B aligned)

__device__ __forceinline__ float block_reduce_256(float v, float* sm) {
  int t = threadIdx.x;
  sm[t] = v;
  __syncthreads();
  for (int s = 128; s > 0; s >>= 1) {
    if (t < s) sm[t] += sm[t + s];
    __syncthreads();
  }
  float r = sm[0];
  __syncthreads();
  return r;
}

// ------------------------------------------------------------------
// 0. zero the accumulation scratch (cnt planes + S matrices)
// ------------------------------------------------------------------
__global__ void imput_init_kernel(float* zbase) {
  int idx = blockIdx.x * 256 + threadIdx.x;   // 7168 floats total
  if (idx < 7168) zbase[idx] = 0.0f;
}

// ------------------------------------------------------------------
// 1. per-locus pass: labels, indicator planes, CE partials, plogp
// ------------------------------------------------------------------
__global__ void imput_locus_kernel(const float* __restrict__ logits,
                                   const float* __restrict__ prob,
                                   const float* __restrict__ y_true,
                                   float* __restrict__ cepart,
                                   float* __restrict__ plogp_out,
                                   unsigned char* __restrict__ labels,
                                   _Float16* __restrict__ Xind) {
  int idx = blockIdx.x * 256 + threadIdx.x;   // 0 .. B*L-1 exact
  const float* y = y_true + (size_t)idx * 5;
  int lbl = 0;
  float best = y[0];
  #pragma unroll
  for (int c = 1; c < 5; ++c) {
    float v = y[c];
    if (v > best) { best = v; lbl = c; }
  }
  bool valid = (lbl != 4);
  labels[idx] = (unsigned char)lbl;
  #pragma unroll
  for (int c = 0; c < 4; ++c)
    Xind[(size_t)c * BL_ + idx] = (_Float16)((lbl == c) ? 1.0f : 0.0f);
  Xind[(size_t)4 * BL_ + idx] = (_Float16)(valid ? 1.0f : 0.0f);

  // cross entropy: lse - logit[label] for valid loci
  float4 lg = reinterpret_cast<const float4*>(logits)[idx];
  float mx = fmaxf(fmaxf(lg.x, lg.y), fmaxf(lg.z, lg.w));
  float se = __expf(lg.x - mx) + __expf(lg.y - mx) + __expf(lg.z - mx) + __expf(lg.w - mx);
  float lse = mx + __logf(se);
  float pick = (lbl == 0) ? lg.x : (lbl == 1) ? lg.y : (lbl == 2) ? lg.z : lg.w;
  float cep = valid ? (lse - pick) : 0.0f;

  // plogp with JS clamp
  float4 p = reinterpret_cast<const float4*>(prob)[idx];
  float pc, s = 0.0f;
  pc = fmaxf(p.x, JS_EPS_F); s += pc * __logf(pc);
  pc = fmaxf(p.y, JS_EPS_F); s += pc * __logf(pc);
  pc = fmaxf(p.z, JS_EPS_F); s += pc * __logf(pc);
  pc = fmaxf(p.w, JS_EPS_F); s += pc * __logf(pc);
  plogp_out[idx] = s;

  __shared__ float sm[256];
  float tot = block_reduce_256(cep, sm);
  if (threadIdx.x == 0) cepart[blockIdx.x] = tot;
}

// ------------------------------------------------------------------
// 2. minimac r2: one thread per (group, locus)
// ------------------------------------------------------------------
__global__ void imput_r2_kernel(const float* __restrict__ prob,
                                const unsigned char* __restrict__ labels,
                                float* __restrict__ r2part) {
  int idx = blockIdx.x * 256 + threadIdx.x;   // 0 .. 8*L-1 exact
  int g = idx / L_, l = idx % L_;
  int nb = 0, alt = 0;
  float pred = 0.0f;
  #pragma unroll
  for (int s = 0; s < 4; ++s) {
    int b = g * 4 + s;
    int lb = labels[(size_t)b * L_ + l];
    nb += (lb != 4);
    alt += (lb >= 1 && lb <= 3);
    float4 p = reinterpret_cast<const float4*>(prob)[(size_t)b * L_ + l];
    pred += p.y + p.z + p.w;
  }
  pred *= 0.25f;
  float af = (float)alt / fmaxf((float)nb, 1.0f);
  float r2acc = 0.0f;
  bool edge = (af == 0.0f) || (af == 1.0f);
  if (!edge) {
    float denom = fmaxf(af * (1.0f - af), 0.01f);
    float d = pred - af;
    r2acc = (d * d / denom) * (float)nb;   // identical r2 for each non-missing sample
  }
  __shared__ float sm[256];
  float tot = block_reduce_256(r2acc, sm);
  if (threadIdx.x == 0) r2part[blockIdx.x] = tot;
}

// ------------------------------------------------------------------
// 3. WMMA one-hot Gram matrices: cnt[m] = X_m * X_m^T  (32x32, K=8192)
//    one wave per (matrix, 16x16 tile, K-split chunk of 512)
// ------------------------------------------------------------------
__global__ void imput_count_wmma_kernel(const _Float16* __restrict__ X,
                                        float* __restrict__ cnt) {
  int bid = blockIdx.x;
  int ks = bid % KSPLIT; bid /= KSPLIT;
  int tile = bid & 3;    bid >>= 2;
  int mat = bid;                       // 0..4
  int ti = tile >> 1, tj = tile & 1;
  int lane = threadIdx.x;
  bool hi = lane >= 16;
  int lr = lane & 15;

  const _Float16* Xm = X + (size_t)mat * BL_;
  const _Float16* rowA = Xm + (size_t)(ti * 16 + lr) * L_;   // A row (M index)
  const _Float16* rowB = Xm + (size_t)(tj * 16 + lr) * L_;   // B column (N index)

  v8f c = {};
  int l0 = ks * (L_ / KSPLIT);
  for (int it = 0; it < (L_ / KSPLIT); it += 32) {
    int base = l0 + it;
    // A operand, contiguous k order chosen here: lane<16 holds K 0..15; lane>=16 holds K 16..31 (matches B)
    v8h a_lo = *reinterpret_cast<const v8h*>(rowA + base + (hi ? 16 : 0));
    v8h a_hi = *reinterpret_cast<const v8h*>(rowA + base + 8 + (hi ? 16 : 0));
    v16h a;
    #pragma unroll
    for (int t = 0; t < 8; ++t) { a[t] = a_lo[t]; a[8 + t] = a_hi[t]; }
    // B operand, same contiguous k order as A: lane<16 holds K 0..15 of column N=lane; lane>=16 holds K 16..31
    v16h b = *reinterpret_cast<const v16h*>(rowB + base + (hi ? 16 : 0));
    c = __builtin_amdgcn_wmma_f32_16x16x32_f16(false, a, false, b, (short)0, c, false, false);
  }
  // C/D f32 16x16 layout: VGPR r, lanes 0-15 -> (M=r, N=lane); lanes 16-31 -> (M=8+r, N=lane-16)
  float* out = cnt + mat * 1024;
  #pragma unroll
  for (int r = 0; r < 8; ++r) {
    int m = r + (hi ? 8 : 0);
    int gi = ti * 16 + m, gj = tj * 16 + lr;
    atomicAdd(&out[gi * 32 + gj], c[r]);   // exact integer-valued adds
  }
}

// ------------------------------------------------------------------
// 4. pairwise JS sums over loci: one block per (i<j) pair
//    js = 0.5*(plogp_i + plogp_j) - sum_c m*log(m),  m = 0.5*(pc_i + pc_j)
// ------------------------------------------------------------------
__global__ void imput_pair_kernel(const float* __restrict__ prob,
                                  const unsigned char* __restrict__ labels,
                                  const float* __restrict__ plogp,
                                  float* __restrict__ S_same,
                                  float* __restrict__ S_diff) {
  int rem = blockIdx.x;     // 0..495
  int i = 0;
  while (rem >= 31 - i) { rem -= 31 - i; ++i; }
  int j = i + 1 + rem;

  const unsigned char* Li = labels + (size_t)i * L_;
  const unsigned char* Lj = labels + (size_t)j * L_;
  const float4* Pi = reinterpret_cast<const float4*>(prob) + (size_t)i * L_;
  const float4* Pj = reinterpret_cast<const float4*>(prob) + (size_t)j * L_;
  const float* pli = plogp + (size_t)i * L_;
  const float* plj = plogp + (size_t)j * L_;

  float ss = 0.0f, sd = 0.0f;
  for (int l = threadIdx.x; l < L_; l += 256) {
    int a = Li[l], b = Lj[l];
    if (a == 4 || b == 4) continue;
    float4 pi4 = Pi[l];
    float4 pj4 = Pj[l];
    float m0 = 0.5f * (fmaxf(pi4.x, JS_EPS_F) + fmaxf(pj4.x, JS_EPS_F));
    float m1 = 0.5f * (fmaxf(pi4.y, JS_EPS_F) + fmaxf(pj4.y, JS_EPS_F));
    float m2 = 0.5f * (fmaxf(pi4.z, JS_EPS_F) + fmaxf(pj4.z, JS_EPS_F));
    float m3 = 0.5f * (fmaxf(pi4.w, JS_EPS_F) + fmaxf(pj4.w, JS_EPS_F));
    float smm = m0 * __logf(m0) + m1 * __logf(m1) + m2 * __logf(m2) + m3 * __logf(m3);
    float js = 0.5f * (pli[l] + plj[l]) - smm;
    if (a == b) ss += js; else sd += js;
  }
  __shared__ float sm[256];
  float tss = block_reduce_256(ss, sm);
  float tsd = block_reduce_256(sd, sm);
  if (threadIdx.x == 0) {
    S_same[i * 32 + j] = tss; S_same[j * 32 + i] = tss;   // symmetric; diag stays 0
    S_diff[i * 32 + j] = tsd; S_diff[j * 32 + i] = tsd;
  }
}

// ------------------------------------------------------------------
// 5. finalize: combine CE, r2, evo-JS into the scalar loss
// ------------------------------------------------------------------
__global__ void imput_final_kernel(const float* __restrict__ yevo,
                                   const float* __restrict__ cepart,
                                   const float* __restrict__ r2part,
                                   const float* __restrict__ cnt,
                                   const float* __restrict__ S_same,
                                   const float* __restrict__ S_diff,
                                   float* __restrict__ out) {
  __shared__ float sm[256];
  __shared__ float rs[32];
  int t = threadIdx.x;

  float ce = 0.0f;
  for (int k = t; k < 1024; k += 256) ce += cepart[k];
  ce = block_reduce_256(ce, sm);

  float r2 = r2part[t];
  r2 = block_reduce_256(r2, sm);

  if (t < 32) {
    float s = 0.0f;
    for (int j = 0; j < 32; ++j) s += __expf(-EVO_LAMBDA_F * yevo[t * 32 + j]);
    rs[t] = s;
  }
  __syncthreads();

  float ts = 0.0f, td = 0.0f, scs = 0.0f, scd = 0.0f;
  for (int p = t; p < 1024; p += 256) {
    int i = p >> 5;
    float csum  = cnt[p] + cnt[1024 + p] + cnt[2048 + p] + cnt[3072 + p];
    float cpair = cnt[4096 + p];
    float cdiff = cpair - csum;
    float cs = fmaxf(csum, 1.0f);
    float cd = fmaxf(cdiff, 1.0f);
    float w = __expf(-EVO_LAMBDA_F * yevo[p]) / (rs[i] + EPS_F);
    ts  += w * S_same[p];                    // (w * S/cnt) * cnt == w * S
    td  += (1.0f - w) * (cd - S_diff[p]);    // (1-w)*(1 - S/cnt)*cnt
    scs += cs;
    scd += cd;
  }
  ts  = block_reduce_256(ts, sm);
  td  = block_reduce_256(td, sm);
  scs = block_reduce_256(scs, sm);
  scd = block_reduce_256(scd, sm);

  if (t == 0) {
    float avg_same = ts / fmaxf(scs, 1.0f);
    float avg_diff = td / fmaxf(scd, 1.0f);
    float evo = EVO_WEIGHT_F * 0.5f * (avg_same + avg_diff) * (float)L_;
    if (!(evo == evo) || fabsf(evo) > 3.0e38f) evo = 0.0f;  // isfinite guard
    float r2loss = -r2 / 8.0f;   // R2_WEIGHT * (-sum*4) / B
    out[0] = ce + r2loss + evo;
  }
}

// ------------------------------------------------------------------
extern "C" void kernel_launch(void* const* d_in, const int* in_sizes, int n_in,
                              void* d_out, int out_size, void* d_ws, size_t ws_size,
                              hipStream_t stream) {
  (void)in_sizes; (void)n_in; (void)out_size; (void)ws_size;
  const float* logits = (const float*)d_in[0];
  const float* prob   = (const float*)d_in[1];
  const float* y_true = (const float*)d_in[2];
  const float* y_evo  = (const float*)d_in[3];
  float* out = (float*)d_out;

  char* ws = (char*)d_ws;
  float*         cepart = (float*)(ws + OFF_CEPART);
  float*         r2part = (float*)(ws + OFF_R2PART);
  float*         cnt    = (float*)(ws + OFF_CNT);
  float*         S_same = (float*)(ws + OFF_SSAME);
  float*         S_diff = (float*)(ws + OFF_SDIFF);
  float*         plogp  = (float*)(ws + OFF_PLOGP);
  unsigned char* labels = (unsigned char*)(ws + OFF_LABEL);
  _Float16*      Xind   = (_Float16*)(ws + OFF_XIND);

  imput_init_kernel<<<28, 256, 0, stream>>>(cnt);   // zeros cnt + S_same + S_diff (contiguous)
  imput_locus_kernel<<<BL_ / 256, 256, 0, stream>>>(logits, prob, y_true,
                                                    cepart, plogp, labels, Xind);
  imput_r2_kernel<<<(8 * L_) / 256, 256, 0, stream>>>(prob, labels, r2part);
  imput_count_wmma_kernel<<<5 * 4 * KSPLIT, 32, 0, stream>>>(Xind, cnt);
  imput_pair_kernel<<<496, 256, 0, stream>>>(prob, labels, plogp, S_same, S_diff);
  imput_final_kernel<<<1, 256, 0, stream>>>(y_evo, cepart, r2part, cnt, S_same, S_diff, out);
}
